// Critic_Transformer_18391049961866
// MI455X (gfx1250) — matrix operand order repaired and checked by reading, not compiled, not run on hardware
//
#include <hip/hip_runtime.h>
#include <math.h>
#include <stdint.h>

// ---------------------------------------------------------------------------
// CDNA5 WMMA types
// ---------------------------------------------------------------------------
typedef __attribute__((ext_vector_type(16))) _Float16 v16h;
typedef __attribute__((ext_vector_type(4)))  _Float16 v4h;
typedef __attribute__((ext_vector_type(8)))  float    v8f;

#define D_MODEL 512
#define SEQ     512
#define BATCH   2
#define NROWS   (BATCH * SEQ)      // 1024
#define N_HEAD  8
#define D_HEAD  64
#define D_FF    2048
#define D_CAT   1088               // 128+256+64+512+64+64
#define EPS_LN   1e-5f
#define EPS_ATTN 1e-6f

// ---------------------------------------------------------------------------
// fp32 -> fp16 bulk converter (weights, once per launch). n must be %4.
// ---------------------------------------------------------------------------
__global__ void cvt16_kernel(const float* __restrict__ src, _Float16* __restrict__ dst,
                             int n4) {
  int i = blockIdx.x * blockDim.x + threadIdx.x;
  if (i >= n4) return;
  float4 v = ((const float4*)src)[i];
  v4h h;
  h[0] = (_Float16)v.x; h[1] = (_Float16)v.y;
  h[2] = (_Float16)v.z; h[3] = (_Float16)v.w;
  ((v4h*)dst)[i] = h;
}

// ---------------------------------------------------------------------------
// Embedding gather + concat (scaled by sqrt(emb_size)) -> f16
// ---------------------------------------------------------------------------
__global__ void embed_kernel(const int* __restrict__ x,
                             const float* __restrict__ t0, const float* __restrict__ t1,
                             const float* __restrict__ t2, const float* __restrict__ t3,
                             const float* __restrict__ t4, const float* __restrict__ t5,
                             _Float16* __restrict__ e16) {
  int row = blockIdx.x;                 // 0..1023
  __shared__ int toks[6];
  if (threadIdx.x < 6) toks[threadIdx.x] = x[row * 6 + threadIdx.x];
  __syncthreads();
  for (int c = threadIdx.x; c < D_CAT; c += blockDim.x) {
    float val;
    if      (c < 128)  val = t0[toks[0] * 128 + c]          * 11.313708499f;   // sqrt(128)
    else if (c < 384)  val = t1[toks[1] * 256 + (c - 128)]  * 16.0f;           // sqrt(256)
    else if (c < 448)  val = t2[toks[2] * 64  + (c - 384)]  * 8.0f;            // sqrt(64)
    else if (c < 960)  val = t3[toks[3] * 512 + (c - 448)]  * 22.627416998f;   // sqrt(512)
    else if (c < 1024) val = t4[toks[4] * 128 + (c - 960)]  * 11.313708499f;
    else               val = t5[toks[5] * 128 + (c - 1024)] * 11.313708499f;
    e16[(size_t)row * D_CAT + c] = (_Float16)val;
  }
}

// ---------------------------------------------------------------------------
// WMMA GEMM:  epi(A16[M,K] @ B16[K,N] + bias[N]) -> C (f32, opt) and/or C16
//   act: 0 = none, 1 = exact GELU ; add_pe: sinusoidal PE (N==D_MODEL only)
//   res: optional f32 residual
// M%128==0, N%64==0, K%32==0. Block = 256 threads (8 wave32 waves).
// Tile 128x64; each wave 32x32 = 4 x v_wmma_f32_16x16x32_f16 per K-step.
// A tile staged via GLOBAL_LOAD_ASYNC_TO_LDS_B128 (ASYNCcnt), no conversion.
// ---------------------------------------------------------------------------
#define LDS_K 40   // f16 K-stride: 80B row stride -> 16B aligned, conflict-free frags

__global__ __launch_bounds__(256)
void gemm_wmma(const _Float16* __restrict__ A16, const _Float16* __restrict__ B16,
               const float* __restrict__ bias, const float* __restrict__ res,
               float* __restrict__ C, _Float16* __restrict__ C16,
               int M, int N, int K, int act, int add_pe) {
  __shared__ __align__(16) _Float16 As[128][LDS_K];  // [m][k]
  __shared__ __align__(16) _Float16 Bs[64][LDS_K];   // [n][k] (transposed stage)

  const int tid  = threadIdx.x;
  const int lane = tid & 31;
  const int wave = tid >> 5;           // 0..7
  const int mo   = (wave >> 1) * 32;   // 32-row strip per wave
  const int no   = (wave & 1) * 32;    // 32-col strip per wave
  const int m0   = blockIdx.y * 128;
  const int n0   = blockIdx.x * 64;

  const int lhalf = lane >> 4;         // 0 | 1
  const int l16   = lane & 15;

  v8f acc[4];                          // [aFrag*2 + bFrag]
  #pragma unroll
  for (int t = 0; t < 4; ++t)
    #pragma unroll
    for (int r = 0; r < 8; ++r) acc[t][r] = 0.0f;

  for (int kb = 0; kb < K; kb += 32) {
    // ---- A tile 128x32 f16: async DMA global->LDS, 16B chunks, no ALU ----
    #pragma unroll
    for (int i = 0; i < 2; ++i) {
      int idx = tid + i * 256;                 // 0..511 chunks of 8 f16
      int mm  = idx >> 2;                      // 0..127
      int kkc = (idx & 3) * 8;                 // 0,8,16,24
      const _Float16* gp = A16 + (size_t)(m0 + mm) * K + kb + kkc;
      unsigned ldsa = (unsigned)(uintptr_t)&As[mm][kkc];  // low 32 bits = LDS offset
      asm volatile("global_load_async_to_lds_b128 %0, %1, off"
                   :: "v"(ldsa), "v"(gp) : "memory");
    }
    // ---- B tile 32x64 f16, transposed stage (coalesced b64 global reads) ----
    #pragma unroll
    for (int i = 0; i < 2; ++i) {
      int idx = tid + i * 256;                 // 0..511
      int kk  = idx >> 4;                      // 0..31
      int nn  = (idx & 15) * 4;                // 0,4,...,60
      v4h hb = *(const v4h*)(B16 + (size_t)(kb + kk) * N + n0 + nn);
      Bs[nn + 0][kk] = hb[0];
      Bs[nn + 1][kk] = hb[1];
      Bs[nn + 2][kk] = hb[2];
      Bs[nn + 3][kk] = hb[3];
    }
    asm volatile("s_wait_asynccnt 0" ::: "memory");
    __syncthreads();

    // ---- fragments per ISA 7.12.2 ----
    v16h af0, af1;
    {
      const int r0 = mo + l16;
      const int r1 = mo + 16 + l16;
      const int kA = lhalf * 8;
      #pragma unroll
      for (int j = 0; j < 8; ++j) {
        af0[j]     = As[r0][kA + j];
        af0[8 + j] = As[r0][16 + kA + j];
        af1[j]     = As[r1][kA + j];
        af1[8 + j] = As[r1][16 + kA + j];
      }
    }
    v16h bf0, bf1;
    {
      const int kB = lhalf * 8;
      const int c0 = no + l16;
      const int c1 = no + 16 + l16;
      #pragma unroll
      for (int j = 0; j < 8; ++j) {
        bf0[j]     = Bs[c0][kB + j];
        bf0[8 + j] = Bs[c0][16 + kB + j];
        bf1[j]     = Bs[c1][kB + j];
        bf1[8 + j] = Bs[c1][16 + kB + j];
      }
    }
    acc[0] = __builtin_amdgcn_wmma_f32_16x16x32_f16(false, af0, false, bf0,
                                                    (short)0, acc[0], false, false);
    acc[1] = __builtin_amdgcn_wmma_f32_16x16x32_f16(false, af0, false, bf1,
                                                    (short)0, acc[1], false, false);
    acc[2] = __builtin_amdgcn_wmma_f32_16x16x32_f16(false, af1, false, bf0,
                                                    (short)0, acc[2], false, false);
    acc[3] = __builtin_amdgcn_wmma_f32_16x16x32_f16(false, af1, false, bf1,
                                                    (short)0, acc[3], false, false);
    __syncthreads();
  }

  // ---- epilogue: C/D layout: VGPR r -> M = r + 8*lhalf, N = l16 ----
  #pragma unroll
  for (int p = 0; p < 2; ++p) {
    #pragma unroll
    for (int t = 0; t < 2; ++t) {
      #pragma unroll
      for (int r = 0; r < 8; ++r) {
        int row = m0 + mo + p * 16 + r + 8 * lhalf;
        int col = n0 + no + t * 16 + l16;
        float val = acc[p * 2 + t][r] + bias[col];
        if (act == 1)  // exact GELU
          val = 0.5f * val * (1.0f + erff(val * 0.70710678118f));
        if (add_pe) {
          int s = row & (SEQ - 1);
          float fr = __expf((float)(col & ~1) * (-9.210340371976184f / (float)D_MODEL));
          val += (col & 1) ? __cosf((float)s * fr) : __sinf((float)s * fr);
        }
        if (res) val += res[(size_t)row * N + col];
        if (C)   C[(size_t)row * N + col] = val;
        if (C16) C16[(size_t)row * N + col] = (_Float16)val;
      }
    }
  }
}

// ---------------------------------------------------------------------------
// Causal linear attention scan, one block per (batch, head).
// phi(x) = elu(x)+1 = x>0 ? x+1 : exp(x). Output written as f16 (next GEMM A).
// ---------------------------------------------------------------------------
__global__ __launch_bounds__(256)
void attn_kernel(const float* __restrict__ Q, const float* __restrict__ K,
                 const float* __restrict__ V, _Float16* __restrict__ O16) {
  const int bh = blockIdx.x;            // 0..15
  const int b  = bh >> 3;
  const int hh = bh & 7;
  const int tid = threadIdx.x;

  __shared__ float kv[D_HEAD * D_HEAD];
  __shared__ float ksum[D_HEAD];
  __shared__ float qf[D_HEAD], kf[D_HEAD], vv[D_HEAD], red[D_HEAD];

  for (int i = tid; i < D_HEAD * D_HEAD; i += 256) kv[i] = 0.0f;
  if (tid < D_HEAD) ksum[tid] = 0.0f;
  __syncthreads();

  for (int s = 0; s < SEQ; ++s) {
    const size_t base = (size_t)(b * SEQ + s) * D_MODEL + hh * D_HEAD;
    if (tid < D_HEAD) {
      float q = Q[base + tid];  qf[tid] = (q > 0.0f) ? q + 1.0f : __expf(q);
      float k = K[base + tid];  kf[tid] = (k > 0.0f) ? k + 1.0f : __expf(k);
      vv[tid] = V[base + tid];
    }
    __syncthreads();
    {
      int i0 = tid * 16;
      #pragma unroll
      for (int i = 0; i < 16; ++i) {
        int idx = i0 + i;
        kv[idx] += kf[idx >> 6] * vv[idx & 63];
      }
    }
    if (tid < D_HEAD) ksum[tid] += kf[tid];
    __syncthreads();
    if (tid < D_HEAD) red[tid] = qf[tid] * ksum[tid];
    __syncthreads();
    if (tid < D_HEAD) {
      float z = 0.0f;
      #pragma unroll 8
      for (int d = 0; d < D_HEAD; ++d) z += red[d];
      float num = 0.0f;
      #pragma unroll 8
      for (int d = 0; d < D_HEAD; ++d) num += qf[d] * kv[d * D_HEAD + tid];
      O16[base + tid] = (_Float16)(num / (z + EPS_ATTN));
    }
    __syncthreads();
  }
}

// ---------------------------------------------------------------------------
// out = LayerNorm(xa + xb) * g + b ; optional f16 sibling for next GEMM A.
// ---------------------------------------------------------------------------
__global__ __launch_bounds__(256)
void addln_kernel(const float* __restrict__ xa, const float* __restrict__ xb,
                  const float* __restrict__ g, const float* __restrict__ bb,
                  float* __restrict__ out, _Float16* __restrict__ out16) {
  const int row = blockIdx.x;
  const int tid = threadIdx.x;
  const size_t i0 = (size_t)row * D_MODEL + tid;
  const size_t i1 = i0 + 256;
  float v0 = xa[i0] + (xb ? xb[i0] : 0.0f);
  float v1 = xa[i1] + (xb ? xb[i1] : 0.0f);

  __shared__ float sb[256];
  sb[tid] = v0 + v1;
  __syncthreads();
  for (int o = 128; o > 0; o >>= 1) { if (tid < o) sb[tid] += sb[tid + o]; __syncthreads(); }
  float mean = sb[0] * (1.0f / D_MODEL);
  __syncthreads();
  float d0 = v0 - mean, d1 = v1 - mean;
  sb[tid] = d0 * d0 + d1 * d1;
  __syncthreads();
  for (int o = 128; o > 0; o >>= 1) { if (tid < o) sb[tid] += sb[tid + o]; __syncthreads(); }
  float inv = rsqrtf(sb[0] * (1.0f / D_MODEL) + EPS_LN);
  float y0 = d0 * inv * g[tid] + bb[tid];
  float y1 = d1 * inv * g[tid + 256] + bb[tid + 256];
  out[i0] = y0;
  out[i1] = y1;
  if (out16) { out16[i0] = (_Float16)y0; out16[i1] = (_Float16)y1; }
}

// ---------------------------------------------------------------------------
// Head collapse: U[d] = sum_i sum_t pw_i[d,t] * vw_i[t]
// ---------------------------------------------------------------------------
__global__ void headU_kernel(const float* pw0, const float* pw1, const float* pw2,
                             const float* pw3, const float* pw4, const float* pw5,
                             const float* vw0, const float* vw1, const float* vw2,
                             const float* vw3, const float* vw4, const float* vw5,
                             float* __restrict__ U) {
  int d = blockIdx.x * blockDim.x + threadIdx.x;
  if (d >= D_MODEL) return;
  const float* pws[6] = {pw0, pw1, pw2, pw3, pw4, pw5};
  const float* vws[6] = {vw0, vw1, vw2, vw3, vw4, vw5};
  const int    vs[6]  = {56, 135, 18, 87, 18, 25};
  float acc = 0.0f;
  for (int i = 0; i < 6; ++i) {
    const float* pwp = pws[i] + (size_t)d * vs[i];
    const float* vwp = vws[i];
    float s = 0.0f;
    for (int t = 0; t < vs[i]; ++t) s += pwp[t] * vwp[t];
    acc += s;
  }
  U[d] = acc;
}

__global__ void headC_kernel(const float* pb0, const float* pb1, const float* pb2,
                             const float* pb3, const float* pb4, const float* pb5,
                             const float* vw0, const float* vw1, const float* vw2,
                             const float* vw3, const float* vw4, const float* vw5,
                             const float* vb0, const float* vb1, const float* vb2,
                             const float* vb3, const float* vb4, const float* vb5,
                             float* __restrict__ Cout) {
  const float* pbs[6] = {pb0, pb1, pb2, pb3, pb4, pb5};
  const float* vws[6] = {vw0, vw1, vw2, vw3, vw4, vw5};
  const float* vbs[6] = {vb0, vb1, vb2, vb3, vb4, vb5};
  const int    vs[6]  = {56, 135, 18, 87, 18, 25};
  float c = 0.0f;
  for (int i = 0; i < 6; ++i) {
    float s = 0.0f;
    for (int t = 0; t < vs[i]; ++t) s += pbs[i][t] * vws[i][t];
    c += s + vbs[i][0];
  }
  Cout[0] = c;
}

// out[b] = (mean_s (h[b,s] . U) + C) / 6
__global__ __launch_bounds__(256)
void finred_kernel(const float* __restrict__ h, const float* __restrict__ U,
                   const float* __restrict__ Cptr, float* __restrict__ out) {
  const int b = blockIdx.x;
  const int tid = threadIdx.x;
  float part = 0.0f;
  for (int s = tid; s < SEQ; s += 256) {
    const float* r = h + (size_t)(b * SEQ + s) * D_MODEL;
    float d = 0.0f;
    #pragma unroll 8
    for (int j = 0; j < D_MODEL; ++j) d += r[j] * U[j];
    part += d;
  }
  __shared__ float sb[256];
  sb[tid] = part;
  __syncthreads();
  for (int o = 128; o > 0; o >>= 1) { if (tid < o) sb[tid] += sb[tid + o]; __syncthreads(); }
  if (tid == 0) out[b] = (sb[0] * (1.0f / SEQ) + Cptr[0]) * (1.0f / 6.0f);
}

// ---------------------------------------------------------------------------
// Host-side orchestration
// ---------------------------------------------------------------------------
static inline void launch_gemm(const _Float16* A16, const _Float16* B16,
                               const float* bias, const float* res,
                               float* C, _Float16* C16, int M, int N, int K,
                               int act, int add_pe, hipStream_t stream) {
  dim3 grid(N / 64, M / 128);
  gemm_wmma<<<grid, 256, 0, stream>>>(A16, B16, bias, res, C, C16, M, N, K, act, add_pe);
}

static inline void launch_cvt(const float* src, _Float16* dst, size_t n, hipStream_t s) {
  int n4 = (int)(n / 4);
  cvt16_kernel<<<(n4 + 255) / 256, 256, 0, s>>>(src, dst, n4);
}

extern "C" void kernel_launch(void* const* d_in, const int* in_sizes, int n_in,
                              void* d_out, int out_size, void* d_ws, size_t ws_size,
                              hipStream_t stream) {
  (void)in_sizes; (void)n_in; (void)out_size; (void)ws_size;

  // --- inputs in setup_inputs() dict order ---
  const int*   x    = (const int*)d_in[0];
  const float* t0   = (const float*)d_in[1];
  const float* t1   = (const float*)d_in[2];
  const float* t2   = (const float*)d_in[3];
  const float* t3   = (const float*)d_in[4];
  const float* t4   = (const float*)d_in[5];
  const float* t5   = (const float*)d_in[6];
  const float* in_w = (const float*)d_in[7];
  const float* in_b = (const float*)d_in[8];
  const float* Wq   = (const float*)d_in[9];
  const float* Wk   = (const float*)d_in[10];
  const float* Wv   = (const float*)d_in[11];
  const float* Wo   = (const float*)d_in[12];
  const float* W1   = (const float*)d_in[13];
  const float* W2   = (const float*)d_in[14];
  const float* bq   = (const float*)d_in[15];
  const float* bk   = (const float*)d_in[16];
  const float* bv   = (const float*)d_in[17];
  const float* bo   = (const float*)d_in[18];
  const float* bf1  = (const float*)d_in[19];
  const float* bf2  = (const float*)d_in[20];
  const float* b1n  = (const float*)d_in[21];
  const float* b2n  = (const float*)d_in[22];
  const float* g1   = (const float*)d_in[23];
  const float* g2   = (const float*)d_in[24];
  const float* gf   = (const float*)d_in[25];
  const float* bfn  = (const float*)d_in[26];
  auto pw = [&](int i) { return (const float*)d_in[27 + 4 * i]; };
  auto pb = [&](int i) { return (const float*)d_in[28 + 4 * i]; };
  auto vw = [&](int i) { return (const float*)d_in[29 + 4 * i]; };
  auto vb = [&](int i) { return (const float*)d_in[30 + 4 * i]; };

  // --- workspace layout ---
  const size_t nAct  = (size_t)NROWS * D_MODEL;     // 524288
  const size_t nWqkv = (size_t)12 * D_MODEL * D_MODEL;
  const size_t nWff  = (size_t)12 * D_MODEL * D_FF;

  float* F = (float*)d_ws;                          // f32 region
  float* h   = F;
  float* tq  = h  + nAct;
  float* tk  = tq + nAct;
  float* tv  = tk + nAct;
  float* h2  = tv + nAct;
  float* U   = h2 + nAct;                           // 513

  _Float16* H = (_Float16*)(U + 520);               // f16 region (16B-aligned offset)
  _Float16* e16   = H;
  _Float16* h16   = e16  + (size_t)NROWS * D_CAT;
  _Float16* h216  = h16  + nAct;
  _Float16* ta16  = h216 + nAct;
  _Float16* ff16  = ta16 + nAct;
  _Float16* inw16 = ff16 + (size_t)NROWS * D_FF;
  _Float16* Wq16  = inw16 + (size_t)D_CAT * D_MODEL;
  _Float16* Wk16  = Wq16 + nWqkv;
  _Float16* Wv16  = Wk16 + nWqkv;
  _Float16* Wo16  = Wv16 + nWqkv;
  _Float16* W116  = Wo16 + nWqkv;
  _Float16* W216  = W116 + nWff;

  // 0) one-shot weight conversion fp32 -> fp16
  launch_cvt(in_w, inw16, (size_t)D_CAT * D_MODEL, stream);
  launch_cvt(Wq, Wq16, nWqkv, stream);
  launch_cvt(Wk, Wk16, nWqkv, stream);
  launch_cvt(Wv, Wv16, nWqkv, stream);
  launch_cvt(Wo, Wo16, nWqkv, stream);
  launch_cvt(W1, W116, nWff, stream);
  launch_cvt(W2, W216, nWff, stream);

  // 1) embeddings -> e16
  embed_kernel<<<NROWS, 256, 0, stream>>>(x, t0, t1, t2, t3, t4, t5, e16);
  // 2) input projection + bias + positional encoding -> h (f32) + h16
  launch_gemm(e16, inw16, in_b, nullptr, h, h16, NROWS, D_MODEL, D_CAT, 0, /*pe*/1, stream);

  // 3) transformer layers
  for (int l = 0; l < 12; ++l) {
    const _Float16* Wq_l = Wq16 + (size_t)l * D_MODEL * D_MODEL;
    const _Float16* Wk_l = Wk16 + (size_t)l * D_MODEL * D_MODEL;
    const _Float16* Wv_l = Wv16 + (size_t)l * D_MODEL * D_MODEL;
    const _Float16* Wo_l = Wo16 + (size_t)l * D_MODEL * D_MODEL;
    const _Float16* W1_l = W116 + (size_t)l * D_MODEL * D_FF;
    const _Float16* W2_l = W216 + (size_t)l * D_FF * D_MODEL;

    launch_gemm(h16, Wq_l, bq + l * D_MODEL, nullptr, tq, nullptr, NROWS, D_MODEL, D_MODEL, 0, 0, stream);
    launch_gemm(h16, Wk_l, bk + l * D_MODEL, nullptr, tk, nullptr, NROWS, D_MODEL, D_MODEL, 0, 0, stream);
    launch_gemm(h16, Wv_l, bv + l * D_MODEL, nullptr, tv, nullptr, NROWS, D_MODEL, D_MODEL, 0, 0, stream);
    attn_kernel<<<BATCH * N_HEAD, 256, 0, stream>>>(tq, tk, tv, ta16);
    launch_gemm(ta16, Wo_l, bo + l * D_MODEL, nullptr, tq, nullptr, NROWS, D_MODEL, D_MODEL, 0, 0, stream);
    addln_kernel<<<NROWS, 256, 0, stream>>>(h, tq, g1 + l * D_MODEL, b1n + l * D_MODEL, h2, h216);
    launch_gemm(h216, W1_l, bf1 + l * D_FF, nullptr, nullptr, ff16, NROWS, D_FF, D_MODEL, /*gelu*/1, 0, stream);
    launch_gemm(ff16, W2_l, bf2 + l * D_MODEL, nullptr, tq, nullptr, NROWS, D_MODEL, D_FF, 0, 0, stream);
    addln_kernel<<<NROWS, 256, 0, stream>>>(h2, tq, g2 + l * D_MODEL, b2n + l * D_MODEL, h, h16);
  }

  // 4) final LN -> h2 (f32 only)
  addln_kernel<<<NROWS, 256, 0, stream>>>(h, nullptr, gf, bfn, h2, nullptr);

  // 5) collapsed output heads
  headU_kernel<<<2, 256, 0, stream>>>(pw(0), pw(1), pw(2), pw(3), pw(4), pw(5),
                                      vw(0), vw(1), vw(2), vw(3), vw(4), vw(5), U);
  headC_kernel<<<1, 1, 0, stream>>>(pb(0), pb(1), pb(2), pb(3), pb(4), pb(5),
                                    vw(0), vw(1), vw(2), vw(3), vw(4), vw(5),
                                    vb(0), vb(1), vb(2), vb(3), vb(4), vb(5), U + D_MODEL);
  finred_kernel<<<BATCH, 256, 0, stream>>>(h2, U, U + D_MODEL, (float*)d_out);
}
